// MLA_6889127543054
// MI455X (gfx1250) — compile-verified, matrix operand order checked by reading, not run on hardware
//
#include <hip/hip_runtime.h>
#include <hip/hip_bf16.h>
#include <stdint.h>

// ---------------------------------------------------------------------------
// MLA (DeepSeek-style) forward on MI455X / gfx1250.
// bf16 WMMA everywhere, vectorized ds_load_b128 fragment loads,
// TDM (tensor_load_to_lds, 6-arg toolchain variant) for attention K tiles.
// B=4, S=2048, D=2048, DC=512, DR=64, H=16, DH=128 (head dim 128+64=192).
// ---------------------------------------------------------------------------

typedef __attribute__((ext_vector_type(16))) __bf16        v16bf;
typedef __attribute__((ext_vector_type(8)))  float         v8f;
typedef __attribute__((ext_vector_type(4)))  unsigned int  u32x4;

#define B_  4
#define S_  2048
#define D_  2048
#define DC_ 512
#define DR_ 64
#define H_  16
#define DH_ 128
#define MSZ (B_ * S_)   // 8192 token rows

#if defined(__has_builtin)
#  if __has_builtin(__builtin_amdgcn_tensor_load_to_lds)
#    define USE_TDM 1
#  endif
#endif
#ifndef USE_TDM
#  define USE_TDM 0
#endif

// ------------------------- bf16 bit helpers --------------------------------
__device__ __forceinline__ unsigned short bf16_bits(float f) {
  unsigned u = __builtin_bit_cast(unsigned, f);
  u += 0x7FFFu + ((u >> 16) & 1u);     // round-to-nearest-even
  return (unsigned short)(u >> 16);
}
__device__ __forceinline__ float bits_f32(unsigned short h) {
  unsigned u = ((unsigned)h) << 16;
  return __builtin_bit_cast(float, u);
}

union FragQ { v16bf v; u32x4 q[2]; };

// A-layout fragment (16x32 bf16, ISA 7.12.2). Per lane the 16 elements are
// two contiguous 8-element (16B) runs of one row:
//   K = hl*8 + [0..7]  and  16 + hl*8 + [0..7]
// -> two ds_load_b128 when the tile is row-major and rows are 16B aligned.
__device__ __forceinline__ v16bf load_frag_row(const unsigned short* p, int ld,
                                               int row0, int k0) {
  const int lane = threadIdx.x & 31;
  const int hl = lane >> 4;
  const int m  = lane & 15;
  const unsigned short* base = p + (size_t)(row0 + m) * ld + k0 + hl * 8;
  FragQ f;
  f.q[0] = *(const u32x4*)(base);
  f.q[1] = *(const u32x4*)(base + 16);
  return f.v;
}

// B fragment from a swizzled tile: layout [ntile][lane][16 elems] so each lane
// reads its 32 contiguous bytes -> two ds_load_b128.
__device__ __forceinline__ v16bf load_frag_sw(const unsigned short* sw, int nt) {
  const int lane = threadIdx.x & 31;
  const unsigned short* base = sw + (size_t)(nt * 32 + lane) * 16;
  FragQ f;
  f.q[0] = *(const u32x4*)(base);
  f.q[1] = *(const u32x4*)(base + 8);
  return f.v;
}

// swizzle map for B-tile stores: element (kk, col) of a [32][16*NT] K-major
// tile lands at sw[col>>4][ (col&15) + 16*((kk>>3)&1) ][ (kk&7) | ((kk&16)>>1) ]
__device__ __forceinline__ int sw_elem(int kk) { return (kk & 7) | ((kk & 16) >> 1); }
__device__ __forceinline__ int sw_lane(int kk, int c) { return (c & 15) + (((kk >> 3) & 1) << 4); }

// ------------------------- TDM 2D tile load --------------------------------
#if USE_TDM
typedef __attribute__((ext_vector_type(4))) unsigned int tdm_v4u;
typedef __attribute__((ext_vector_type(8))) int          tdm_v8i;
typedef __attribute__((ext_vector_type(4))) int          tdm_v4i;

// DMA a [tile_h x tile_w] bf16 tile (row stride row_stride elems) from global
// into LDS at byte offset lds_off, rows packed back-to-back.
__device__ __forceinline__ void tdm_load_2d(unsigned lds_off, const void* gptr,
                                            unsigned tile_w, unsigned tile_h,
                                            unsigned long long row_stride) {
  unsigned long long ga = (unsigned long long)gptr;
  tdm_v4u g0;
  g0[0] = 1u;                                              // count=1, user mode
  g0[1] = lds_off;                                         // lds_addr (bytes)
  g0[2] = (unsigned)ga;                                    // global_addr lo
  g0[3] = (unsigned)((ga >> 32) & 0x01FFFFFFull) | (2u << 30); // addr[56:32] | type=2
  tdm_v8i g1;
  g1[0] = (int)(1u << 16);                                 // data_size=1 (2B), wg_mask=0
  g1[1] = (int)((tile_w & 0xFFFFu) << 16);                 // tensor_dim0[15:0]
  g1[2] = (int)(((tile_w >> 16) & 0xFFFFu) | ((tile_h & 0xFFFFu) << 16)); // td0 hi | td1 lo
  g1[3] = (int)(((tile_h >> 16) & 0xFFFFu) | ((tile_w & 0xFFFFu) << 16)); // td1 hi | tile_dim0
  g1[4] = (int)(tile_h & 0xFFFFu);                         // tile_dim1 (tile_dim2=0)
  g1[5] = (int)(unsigned)(row_stride & 0xFFFFFFFFull);     // dim0_stride lo32
  g1[6] = (int)(unsigned)((row_stride >> 32) & 0xFFFFull); // dim0_stride hi16
  g1[7] = 0;
  tdm_v4i gz4 = {0, 0, 0, 0};
  tdm_v8i gz8 = {0, 0, 0, 0, 0, 0, 0, 0};
  // 6-arg toolchain variant: (g0, g1, g2, g3, g4, cpol)
  __builtin_amdgcn_tensor_load_to_lds(g0, g1, gz4, gz4, gz8, 0);
}
#endif

// ---------------------------------------------------------------------------
// Generic GEMM:  C[M,N] = A[M,K] * W[K,N] + bias   (row-major, dense)
// Tile 128x128x32, 256 threads = 8 waves (4 M x 2 N), wave = 32x64 output.
// ---------------------------------------------------------------------------
#define BM 128
#define BN 128
#define BK 32

__global__ __launch_bounds__(256)
void mla_gemm(const void* __restrict__ Aall, const float* __restrict__ W,
              const float* __restrict__ bias, void* __restrict__ Cout,
              int M, int N, int K, int a_bf16, int c_bf16) {
  __shared__ unsigned short As[BM][BK];        // row-major, frag-vectorizable
  __shared__ unsigned short Bsw[8][32][16];    // swizzled fragment-ready

  const int t    = threadIdx.x;
  const int lane = t & 31;
  const int wave = t >> 5;
  const int wm   = wave >> 1;    // 0..3 : M
  const int wn   = wave & 1;     // 0..1 : N
  const int hl   = lane >> 4;
  const int l16  = lane & 15;

  const int row0 = blockIdx.y * BM;
  const int col0 = blockIdx.x * BN;

  const float*          Af = (const float*)Aall;
  const unsigned short* Ab = (const unsigned short*)Aall;

  v8f zero = {};
  v8f acc[2][4];
#pragma unroll
  for (int mi = 0; mi < 2; ++mi)
#pragma unroll
    for (int ni = 0; ni < 4; ++ni) acc[mi][ni] = zero;

  for (int k0 = 0; k0 < K; k0 += BK) {
    // CDNA5 speculative prefetch of next K tile (global_prefetch_b8)
    if (k0 + BK < K) {
      if (a_bf16)
        __builtin_prefetch(&Ab[(size_t)(row0 + (t >> 1)) * K + k0 + BK], 0, 1);
      else
        __builtin_prefetch(&Af[(size_t)(row0 + (t >> 1)) * K + k0 + BK], 0, 1);
      if (col0 + (t & 127) < N)
        __builtin_prefetch(&W[(size_t)(k0 + BK + (t >> 7)) * N + col0 + (t & 127)], 0, 1);
    }

    // A tile 128x32 -> bf16, 32-bit granularity (8 iters)
#pragma unroll
    for (int i = 0; i < (BM * BK) / 512; ++i) {
      int idx = t + i * 256;
      int r = idx >> 4, cw = idx & 15;           // cw: pair index, kk = 2*cw
      unsigned pk;
      if (a_bf16) {
        pk = ((const unsigned*)Ab)[((size_t)(row0 + r) * K + k0) / 2 + cw];
      } else {
        float a0 = Af[(size_t)(row0 + r) * K + k0 + 2 * cw];
        float a1 = Af[(size_t)(row0 + r) * K + k0 + 2 * cw + 1];
        pk = (unsigned)bf16_bits(a0) | ((unsigned)bf16_bits(a1) << 16);
      }
      *(unsigned*)&As[r][2 * cw] = pk;
    }

    // W tile 32x128 -> swizzled bf16, paired kk rows -> one u32 store (8 iters)
#pragma unroll
    for (int i = 0; i < (BK * BN) / 512; ++i) {
      int idx = t + i * 256;
      int c  = idx & 127;
      int kk = (idx >> 7) * 2;                   // even kk
      float w0 = 0.0f, w1 = 0.0f;
      if (col0 + c < N) {
        w0 = W[(size_t)(k0 + kk) * N + col0 + c];
        w1 = W[(size_t)(k0 + kk + 1) * N + col0 + c];
      }
      unsigned pk = (unsigned)bf16_bits(w0) | ((unsigned)bf16_bits(w1) << 16);
      *(unsigned*)&Bsw[c >> 4][sw_lane(kk, c)][sw_elem(kk)] = pk;
    }
    __syncthreads();

    v16bf afrag[2];
#pragma unroll
    for (int mi = 0; mi < 2; ++mi)
      afrag[mi] = load_frag_row(&As[0][0], BK, wm * 32 + mi * 16, 0);
#pragma unroll
    for (int ni = 0; ni < 4; ++ni) {
      v16bf bfrag = load_frag_sw(&Bsw[0][0][0], wn * 4 + ni);
#pragma unroll
      for (int mi = 0; mi < 2; ++mi)
        acc[mi][ni] = __builtin_amdgcn_wmma_f32_16x16x32_bf16(
            false, afrag[mi], false, bfrag, (short)0, acc[mi][ni], false, false);
    }
    __syncthreads();
  }

  // epilogue: C/D layout row = r + 8*hl, col = lane&15
#pragma unroll
  for (int mi = 0; mi < 2; ++mi)
#pragma unroll
    for (int ni = 0; ni < 4; ++ni)
#pragma unroll
      for (int r = 0; r < 8; ++r) {
        int row = row0 + wm * 32 + mi * 16 + r + hl * 8;
        int col = col0 + wn * 64 + ni * 16 + l16;
        if (col < N) {
          float vv = acc[mi][ni][r] + bias[col];
          if (c_bf16)
            ((unsigned short*)Cout)[(size_t)row * N + col] = bf16_bits(vv);
          else
            ((float*)Cout)[(size_t)row * N + col] = vv;
        }
      }
}

// ---------------------------------------------------------------------------
// RoPE (interleaved rotate-half), f32 in -> bf16 bits out. x: [MSZ, 64]
// ---------------------------------------------------------------------------
__global__ __launch_bounds__(256)
void mla_rope(const float* __restrict__ x, unsigned short* __restrict__ out) {
  int i = blockIdx.x * 256 + threadIdx.x;
  if (i >= MSZ * DR_) return;
  int row = i >> 6;
  int d   = i & 63;
  int pos = row & (S_ - 1);
  float idx2 = (float)((d >> 1) << 1);
  float inv_freq = __powf(10000.0f, -idx2 / (float)DR_);
  float ang = (float)pos * inv_freq;
  float c = __cosf(ang), s = __sinf(ang);
  float xv = x[i];
  float xr = (d & 1) ? x[i - 1] : -x[i + 1];
  out[i] = bf16_bits(xv * c + xr * s);
}

// ---------------------------------------------------------------------------
// Flash attention (causal), bf16 WMMA, online softmax.
// grid = (S/64, B*H), block = 128 (4 waves). 64 queries per block, key tiles
// of 32. Q/K: 128 nope + 64 rope dims (split arrays so TDM tiles are
// contiguous). V swizzled into fragment-ready layout.
// ---------------------------------------------------------------------------
__global__ __launch_bounds__(128)
void mla_flash_attn(const unsigned short* __restrict__ q_nope,
                    const unsigned short* __restrict__ q_rope,
                    const unsigned short* __restrict__ k_nope,
                    const unsigned short* __restrict__ k_rope,
                    const unsigned short* __restrict__ vmat,
                    unsigned short* __restrict__ out) {
  __shared__ unsigned short Qsn[64][128];
  __shared__ unsigned short Qsr[64][64];
  __shared__ unsigned short Ksn[32][128];
  __shared__ unsigned short Ksr[32][64];
  __shared__ unsigned short Vsw[8][32][16];   // swizzled B tiles (128 V dims)
  __shared__ unsigned short Ps[64][32];

  const int t    = threadIdx.x;
  const int lane = t & 31;
  const int w    = t >> 5;     // wave 0..3
  const int hl   = lane >> 4;
  const int l16  = lane & 15;

  const int q0 = blockIdx.x * 64;
  const int b  = blockIdx.y >> 4;
  const int h  = blockIdx.y & 15;

  const float SCALE = 0.07216878364870323f;  // 1/sqrt(192)

  // ---- load Q tile (64 rows), 32-bit granularity
  {
    const unsigned* qn32 = (const unsigned*)q_nope;
    const unsigned* qr32 = (const unsigned*)q_rope;
    for (int idx = t; idx < 64 * 64; idx += 128) {          // Qsn: 64 u32/row
      int r = idx >> 6, cw = idx & 63;
      size_t srow = (size_t)b * S_ + q0 + r;
      ((unsigned*)&Qsn[r][0])[cw] = qn32[(srow * D_ + (size_t)h * DH_) / 2 + cw];
    }
    for (int idx = t; idx < 64 * 32; idx += 128) {          // Qsr: 32 u32/row
      int r = idx >> 5, cw = idx & 31;
      size_t srow = (size_t)b * S_ + q0 + r;
      ((unsigned*)&Qsr[r][0])[cw] = qr32[(srow * DR_) / 2 + cw];
    }
  }
  __syncthreads();

  v8f zero = {};
  float m_i[8], l_i[8];
  v8f o[8];
#pragma unroll
  for (int r = 0; r < 8; ++r) { m_i[r] = -3.0e38f; l_i[r] = 0.0f; }
#pragma unroll
  for (int no = 0; no < 8; ++no) o[no] = zero;

  for (int k0 = 0; k0 <= q0 + 32; k0 += 32) {
#if USE_TDM
    // ---- K tiles via Tensor Data Mover (async DMA to LDS), wave 0 issues
    if (w == 0) {
      tdm_load_2d((unsigned)(size_t)&Ksn[0][0],
                  k_nope + ((size_t)b * S_ + k0) * D_ + (size_t)h * DH_,
                  DH_, 32, (unsigned long long)D_);
      tdm_load_2d((unsigned)(size_t)&Ksr[0][0],
                  k_rope + ((size_t)b * S_ + k0) * DR_,
                  DR_, 32, (unsigned long long)DR_);
    }
#else
    {
      const unsigned* kn32 = (const unsigned*)k_nope;
      const unsigned* kr32 = (const unsigned*)k_rope;
      for (int idx = t; idx < 32 * 64; idx += 128) {
        int r = idx >> 6, cw = idx & 63;
        size_t srow = (size_t)b * S_ + k0 + r;
        ((unsigned*)&Ksn[r][0])[cw] = kn32[(srow * D_ + (size_t)h * DH_) / 2 + cw];
      }
      for (int idx = t; idx < 32 * 32; idx += 128) {
        int r = idx >> 5, cw = idx & 31;
        size_t srow = (size_t)b * S_ + k0 + r;
        ((unsigned*)&Ksr[r][0])[cw] = kr32[(srow * DR_) / 2 + cw];
      }
    }
#endif
    // ---- V tile 32x128 -> swizzled (paired key rows -> one u32 store)
    for (int idx = t; idx < (32 * 128) / 2; idx += 128) {
      int c  = idx & 127;
      int kk = (idx >> 7) * 2;
      size_t srow = (size_t)b * S_ + k0 + kk;
      unsigned v0 = vmat[srow * D_ + (size_t)h * DH_ + c];
      unsigned v1 = vmat[(srow + 1) * D_ + (size_t)h * DH_ + c];
      *(unsigned*)&Vsw[c >> 4][sw_lane(kk, c)][sw_elem(kk)] = v0 | (v1 << 16);
    }
#if USE_TDM
    if (w == 0) __builtin_amdgcn_s_wait_tensorcnt(0);
#endif
    __syncthreads();

    // ---- S = Q * K^T : wave computes 16x32, K-dim 192 in 6 chunks
    v8f sacc[2];
    sacc[0] = zero; sacc[1] = zero;
#pragma unroll
    for (int kc = 0; kc < 6; ++kc) {
      const unsigned short* qsrc = (kc < 4) ? &Qsn[0][0] : &Qsr[0][0];
      const unsigned short* ksrc = (kc < 4) ? &Ksn[0][0] : &Ksr[0][0];
      int ld = (kc < 4) ? 128 : 64;
      int ko = (kc < 4) ? kc * 32 : (kc - 4) * 32;
      v16bf a = load_frag_row(qsrc, ld, w * 16, ko);
#pragma unroll
      for (int ni = 0; ni < 2; ++ni) {
        v16bf kb = load_frag_row(ksrc, ld, ni * 16, ko);  // transposed-B == A-layout
        sacc[ni] = __builtin_amdgcn_wmma_f32_16x16x32_bf16(
            false, a, false, kb, (short)0, sacc[ni], false, false);
      }
    }

    // ---- online softmax per row (row = r + 8*hl within wave's 16 rows)
#pragma unroll
    for (int r = 0; r < 8; ++r) {
      int qpos = q0 + w * 16 + r + hl * 8;
      float mx = -3.0e38f;
#pragma unroll
      for (int ni = 0; ni < 2; ++ni) {
        int kpos = k0 + ni * 16 + l16;
        float sv = sacc[ni][r] * SCALE;
        sv = (kpos > qpos) ? -10000.0f : sv;   // causal mask (matches ref)
        sacc[ni][r] = sv;
        mx = fmaxf(mx, sv);
      }
#pragma unroll
      for (int dd = 1; dd <= 8; dd <<= 1)
        mx = fmaxf(mx, __shfl_xor(mx, dd, 32));
      float mn   = fmaxf(m_i[r], mx);
      float corr = __expf(m_i[r] - mn);
      float rs = 0.0f;
#pragma unroll
      for (int ni = 0; ni < 2; ++ni) {
        float p = __expf(sacc[ni][r] - mn);
        Ps[w * 16 + r + hl * 8][ni * 16 + l16] = bf16_bits(p);
        rs += p;
      }
#pragma unroll
      for (int dd = 1; dd <= 8; dd <<= 1) rs += __shfl_xor(rs, dd, 32);
      l_i[r] = l_i[r] * corr + rs;
      m_i[r] = mn;
#pragma unroll
      for (int no = 0; no < 8; ++no) o[no][r] *= corr;
    }

    // ---- O += P (16x32) * V (32x128)
    v16bf pf = load_frag_row(&Ps[0][0], 32, w * 16, 0);
#pragma unroll
    for (int no = 0; no < 8; ++no) {
      v16bf vf = load_frag_sw(&Vsw[0][0][0], no);
      o[no] = __builtin_amdgcn_wmma_f32_16x16x32_bf16(
          false, pf, false, vf, (short)0, o[no], false, false);
    }
    __syncthreads();
  }

  // ---- normalize and write bf16 attn output in [B,S,D] layout
#pragma unroll
  for (int no = 0; no < 8; ++no)
#pragma unroll
    for (int r = 0; r < 8; ++r) {
      int row = q0 + w * 16 + r + hl * 8;
      int col = h * DH_ + no * 16 + l16;
      out[((size_t)b * S_ + row) * D_ + col] = bf16_bits(o[no][r] / l_i[r]);
    }
}

// ---------------------------------------------------------------------------
// Host: workspace layout + launch sequence
// ---------------------------------------------------------------------------
extern "C" void kernel_launch(void* const* d_in, const int* in_sizes, int n_in,
                              void* d_out, int out_size, void* d_ws, size_t ws_size,
                              hipStream_t stream) {
  (void)in_sizes; (void)n_in; (void)out_size; (void)ws_size;

  const float* q    = (const float*)d_in[0];
  const float* kv   = (const float*)d_in[1];
  // d_in[2] = causal mask, applied analytically in the attention kernel
  const float* Wdq  = (const float*)d_in[3];
  const float* bdq  = (const float*)d_in[4];
  const float* Wdkv = (const float*)d_in[5];
  const float* bdkv = (const float*)d_in[6];
  const float* Wqr  = (const float*)d_in[7];
  const float* bqr  = (const float*)d_in[8];
  const float* Wkr  = (const float*)d_in[9];
  const float* bkr  = (const float*)d_in[10];
  const float* Wuq  = (const float*)d_in[11];
  const float* buq  = (const float*)d_in[12];
  const float* Wuk  = (const float*)d_in[13];
  const float* buk  = (const float*)d_in[14];
  const float* Wuv  = (const float*)d_in[15];
  const float* buv  = (const float*)d_in[16];
  const float* Wo   = (const float*)d_in[17];
  const float* bo   = (const float*)d_in[18];
  float* out = (float*)d_out;

  char* ws = (char*)d_ws;
  size_t off = 0;
  float* c_q  = (float*)(ws + off); off += (size_t)MSZ * DC_ * sizeof(float);
  float* c_kv = (float*)(ws + off); off += (size_t)MSZ * DC_ * sizeof(float);
  unsigned short* q_nope = (unsigned short*)(ws + off); off += (size_t)MSZ * D_ * 2;
  unsigned short* k_nope = (unsigned short*)(ws + off); off += (size_t)MSZ * D_ * 2;
  unsigned short* v_b    = (unsigned short*)(ws + off); off += (size_t)MSZ * D_ * 2;
  float* qr_lin = (float*)(ws + off); off += (size_t)MSZ * DR_ * sizeof(float);
  float* kr_lin = (float*)(ws + off); off += (size_t)MSZ * DR_ * sizeof(float);
  unsigned short* q_r  = (unsigned short*)(ws + off); off += (size_t)MSZ * DR_ * 2;
  unsigned short* k_r  = (unsigned short*)(ws + off); off += (size_t)MSZ * DR_ * 2;
  unsigned short* attn = (unsigned short*)(ws + off); off += (size_t)MSZ * D_ * 2;

  dim3 blk(256);
#define GRID2(Mv, Nv) dim3((unsigned)(((Nv) + BN - 1) / BN), (unsigned)(((Mv) + BM - 1) / BM))

  // down-projections (f32 out)
  mla_gemm<<<GRID2(MSZ, DC_), blk, 0, stream>>>(q,  Wdq,  bdq,  c_q,  MSZ, DC_, D_, 0, 0);
  mla_gemm<<<GRID2(MSZ, DC_), blk, 0, stream>>>(kv, Wdkv, bdkv, c_kv, MSZ, DC_, D_, 0, 0);

  // up-projections (bf16 out, consumed by attention)
  mla_gemm<<<GRID2(MSZ, D_), blk, 0, stream>>>(c_q,  Wuq, buq, q_nope, MSZ, D_, DC_, 0, 1);
  mla_gemm<<<GRID2(MSZ, D_), blk, 0, stream>>>(c_kv, Wuk, buk, k_nope, MSZ, D_, DC_, 0, 1);
  mla_gemm<<<GRID2(MSZ, D_), blk, 0, stream>>>(c_kv, Wuv, buv, v_b,    MSZ, D_, DC_, 0, 1);

  // rope branch (N=64 GEMMs, f32 out -> RoPE -> bf16)
  mla_gemm<<<GRID2(MSZ, DR_), blk, 0, stream>>>(c_q, Wqr, bqr, qr_lin, MSZ, DR_, DC_, 0, 0);
  mla_gemm<<<GRID2(MSZ, DR_), blk, 0, stream>>>(kv,  Wkr, bkr, kr_lin, MSZ, DR_, D_,  0, 0);
  mla_rope<<<(MSZ * DR_ + 255) / 256, 256, 0, stream>>>(qr_lin, q_r);
  mla_rope<<<(MSZ * DR_ + 255) / 256, 256, 0, stream>>>(kr_lin, k_r);

  // causal flash attention, bf16 out in [B,S,D]
  mla_flash_attn<<<dim3(S_ / 64, B_ * H_), 128, 0, stream>>>(
      q_nope, q_r, k_nope, k_r, v_b, attn);

  // output projection (bf16 A, f32 out)
  mla_gemm<<<GRID2(MSZ, D_), blk, 0, stream>>>(attn, Wo, bo, out, MSZ, D_, D_, 1, 0);
}
